// MultiHeadSelfAttentionLayer_42082089566362
// MI455X (gfx1250) — matrix operand order repaired and checked by reading, not compiled, not run on hardware
//
#include <hip/hip_runtime.h>

// ---------------------------------------------------------------------------
// MHA forward for MI455X (gfx1250, wave32, WMMA).
// B=2, S=2048, DIM=512, H=8, DEPTH=64. Outputs: out [B,S,DIM] f32 then
// attn [B,H,S,S] f32, concatenated in d_out.
//
// HBM-bound on the 256MB attn tensor (~1.07GB total traffic ~46us @23.3TB/s).
// All matmuls use v_wmma_f32_16x16x32_bf16 (f32 accumulate) on bf16-converted
// operands staged through LDS. f32->bf16 staging packs pairs into dwords so
// LDS writes are ds_store_b64 instead of 4x ds_store_b16.
// ---------------------------------------------------------------------------

#define BATCH 2
#define SEQ   2048
#define DIMSZ 512
#define HEADS 8
#define DEPTH 64

typedef __attribute__((ext_vector_type(16))) __bf16 v16bf;
typedef __attribute__((ext_vector_type(8)))  float  v8f;

union FragBits {
    v16bf v;
    uint4 q[2];
};

__device__ __forceinline__ unsigned short f32_to_bf16_bits(float f) {
    unsigned int u = __float_as_uint(f);
    u += 0x7FFFu + ((u >> 16) & 1u);   // round-to-nearest-even
    return (unsigned short)(u >> 16);
}

// Pack two floats into one dword holding two bf16 (lo in [15:0], hi in [31:16]).
__device__ __forceinline__ unsigned int f32x2_to_bf16x2(float lo, float hi) {
    unsigned int ulo = __float_as_uint(lo);
    unsigned int uhi = __float_as_uint(hi);
    ulo += 0x7FFFu + ((ulo >> 16) & 1u);
    uhi += 0x7FFFu + ((uhi >> 16) & 1u);
    return (ulo >> 16) | (uhi & 0xFFFF0000u);
}

// A fragment (16x32 bf16, MxK): lane row = lane&15; k = (lane>>4)*16 + j (j=0..15)
// -> two contiguous 16B LDS reads per lane from a row-major [.,ldk] tile.
__device__ __forceinline__ v16bf load_a_frag(const unsigned short* lds, int ldk,
                                             int row, int kk, int lane) {
    const uint4* p = (const uint4*)(lds + row * ldk + kk);
    int g = (lane >> 4) * 2;      // k-base 0 or 16 shorts -> uint4 index 0/2
    FragBits f;
    f.q[0] = p[g];                // k = base + 0..7
    f.q[1] = p[g + 1];            // k = base + 8..15
    return f.v;
}

// B fragment (32x16 bf16, KxN): lane col = lane&15; k = (lane>>4)*16 + j (j=0..15)
// -> one contiguous 32B LDS read per lane from an [n][ldk] (k-contiguous) tile.
__device__ __forceinline__ v16bf load_b_frag(const unsigned short* lds, int ldk,
                                             int col, int kk, int lane) {
    const uint4* p = (const uint4*)(lds + col * ldk + kk);
    int g = (lane >> 4) * 2;      // k-base 0 or 16 shorts -> uint4 index 0/2
    FragBits f;
    f.q[0] = p[g];
    f.q[1] = p[g + 1];
    return f.v;
}

__device__ __forceinline__ v8f wmma_bf16(v16bf a, v16bf b, v8f c) {
    return __builtin_amdgcn_wmma_f32_16x16x32_bf16(
        /*neg_a=*/false, a, /*neg_b=*/false, b,
        /*c_mod=*/(short)0, c, /*reuse_a=*/false, /*reuse_b=*/false);
}

// ---------------------------------------------------------------------------
// Kernel 1: Q/K/V projection. Y = X @ W + b, stored bf16 head-major [B,H,S,64].
// Tile: 128(M) x 64(N), BK=64, 256 threads = 8 waves, wave w owns rows w*16..w*16+15.
// ---------------------------------------------------------------------------
__global__ __launch_bounds__(256) void qkv_gemm_kernel(
    const float* __restrict__ X,
    const float* __restrict__ Wq, const float* __restrict__ Wk, const float* __restrict__ Wv,
    const float* __restrict__ bq, const float* __restrict__ bk, const float* __restrict__ bv,
    unsigned short* __restrict__ Qh, unsigned short* __restrict__ Kh,
    unsigned short* __restrict__ Vh)
{
    __shared__ alignas(16) unsigned short lA[128 * 64];  // X tile   [m][k] bf16
    __shared__ alignas(16) unsigned short lB[64 * 64];   // W tile   [n][k] bf16 (transposed)

    const int which = blockIdx.z;
    const float* W    = (which == 0) ? Wq : (which == 1) ? Wk : Wv;
    const float* bias = (which == 0) ? bq : (which == 1) ? bk : bv;
    unsigned short* Out = (which == 0) ? Qh : (which == 1) ? Kh : Vh;

    const int m0 = blockIdx.y * 128;
    const int n0 = blockIdx.x * 64;
    const int tid = threadIdx.x, lane = tid & 31, wave = tid >> 5;

    v8f acc[4] = {};

    for (int k0 = 0; k0 < DIMSZ; k0 += 64) {
        // A: 128x64 f32 -> bf16 (8192 elems; float4 read -> packed uint2 LDS store)
        #pragma unroll
        for (int it = 0; it < 8; ++it) {
            int idx = (it * 256 + tid) * 4;
            int r = idx >> 6, c = idx & 63;
            float4 f = *(const float4*)&X[(size_t)(m0 + r) * DIMSZ + k0 + c];
            uint2 pk;
            pk.x = f32x2_to_bf16x2(f.x, f.y);
            pk.y = f32x2_to_bf16x2(f.z, f.w);
            *(uint2*)&lA[r * 64 + c] = pk;
        }
        // B: W[k][n] 64x64 f32 -> transpose into lB[n][k] bf16
        #pragma unroll
        for (int it = 0; it < 4; ++it) {
            int idx = (it * 256 + tid) * 4;
            int kk = idx >> 6, nn = idx & 63;
            float4 f = *(const float4*)&W[(size_t)(k0 + kk) * DIMSZ + n0 + nn];
            lB[(nn + 0) * 64 + kk] = f32_to_bf16_bits(f.x);
            lB[(nn + 1) * 64 + kk] = f32_to_bf16_bits(f.y);
            lB[(nn + 2) * 64 + kk] = f32_to_bf16_bits(f.z);
            lB[(nn + 3) * 64 + kk] = f32_to_bf16_bits(f.w);
        }
        __syncthreads();

        const int arow = wave * 16 + (lane & 15);
        #pragma unroll
        for (int ks = 0; ks < 64; ks += 32) {
            v16bf a = load_a_frag(lA, 64, arow, ks, lane);
            #pragma unroll
            for (int nt = 0; nt < 4; ++nt) {
                v16bf b = load_b_frag(lB, 64, nt * 16 + (lane & 15), ks, lane);
                acc[nt] = wmma_bf16(a, b, acc[nt]);
            }
        }
        __syncthreads();
    }

    // Epilogue: scatter bf16 into [B,H,S,DEPTH]
    const int half = lane >> 4;
    #pragma unroll
    for (int nt = 0; nt < 4; ++nt) {
        int n = n0 + nt * 16 + (lane & 15);
        int h = n >> 6, d = n & 63;
        float bval = bias[n];
        #pragma unroll
        for (int i = 0; i < 8; ++i) {
            int m  = m0 + wave * 16 + half * 8 + i;
            int bb = m >> 11, ss = m & 2047;
            Out[(((size_t)bb * HEADS + h) * SEQ + ss) * DEPTH + d] =
                f32_to_bf16_bits(acc[nt][i] + bval);
        }
    }
}

// ---------------------------------------------------------------------------
// Kernel 2: logits = Q.K^T / 8 + mask*(-1e9), f32 into attn region of d_out.
// K tiles are already in B-fragment-friendly layout ([s_k][d], d contiguous).
// ---------------------------------------------------------------------------
__global__ __launch_bounds__(256) void logits_kernel(
    const unsigned short* __restrict__ Qh, const unsigned short* __restrict__ Kh,
    const float* __restrict__ mask, float* __restrict__ attn)
{
    __shared__ alignas(16) unsigned short lQ[128 * 64];
    __shared__ alignas(16) unsigned short lK[64 * 64];

    const int bh = blockIdx.z;            // b*H + h
    const int q0 = blockIdx.y * 128;
    const int k0 = blockIdx.x * 64;
    const int tid = threadIdx.x, lane = tid & 31, wave = tid >> 5;

    const unsigned short* Qbase = Qh + (size_t)bh * SEQ * DEPTH + (size_t)q0 * DEPTH;
    const unsigned short* Kbase = Kh + (size_t)bh * SEQ * DEPTH + (size_t)k0 * DEPTH;

    #pragma unroll
    for (int it = 0; it < 4; ++it)
        ((uint4*)lQ)[it * 256 + tid] = ((const uint4*)Qbase)[it * 256 + tid];
    #pragma unroll
    for (int it = 0; it < 2; ++it)
        ((uint4*)lK)[it * 256 + tid] = ((const uint4*)Kbase)[it * 256 + tid];
    __syncthreads();

    v8f acc[4] = {};
    const int arow = wave * 16 + (lane & 15);
    #pragma unroll
    for (int ks = 0; ks < 64; ks += 32) {
        v16bf a = load_a_frag(lQ, 64, arow, ks, lane);
        #pragma unroll
        for (int nt = 0; nt < 4; ++nt) {
            v16bf b = load_b_frag(lK, 64, nt * 16 + (lane & 15), ks, lane);
            acc[nt] = wmma_bf16(a, b, acc[nt]);
        }
    }

    const int bb = bh >> 3;
    const int half = lane >> 4;
    #pragma unroll
    for (int nt = 0; nt < 4; ++nt) {
        int n = k0 + nt * 16 + (lane & 15);
        float mterm = mask[(size_t)bb * SEQ + n] * -1e9f;
        #pragma unroll
        for (int i = 0; i < 8; ++i) {
            int m = q0 + wave * 16 + half * 8 + i;
            attn[((size_t)bh * SEQ + m) * SEQ + n] = acc[nt][i] * 0.125f + mterm;
        }
    }
}

// ---------------------------------------------------------------------------
// Kernel 3: in-place row softmax over attn (rows of length 2048).
// ---------------------------------------------------------------------------
__global__ __launch_bounds__(256) void softmax_kernel(float* __restrict__ attn)
{
    float* p = attn + (size_t)blockIdx.x * SEQ;
    const int tid = threadIdx.x;
    __shared__ float red[256];

    float vals[8];
    float m = -1e30f;
    #pragma unroll
    for (int i = 0; i < 8; ++i) { vals[i] = p[tid + i * 256]; m = fmaxf(m, vals[i]); }
    red[tid] = m; __syncthreads();
    for (int s = 128; s > 0; s >>= 1) {
        if (tid < s) red[tid] = fmaxf(red[tid], red[tid + s]);
        __syncthreads();
    }
    m = red[0]; __syncthreads();

    float sum = 0.f;
    #pragma unroll
    for (int i = 0; i < 8; ++i) { vals[i] = __expf(vals[i] - m); sum += vals[i]; }
    red[tid] = sum; __syncthreads();
    for (int s = 128; s > 0; s >>= 1) {
        if (tid < s) red[tid] += red[tid + s];
        __syncthreads();
    }
    float inv = 1.0f / red[0];
    #pragma unroll
    for (int i = 0; i < 8; ++i) p[tid + i * 256] = vals[i] * inv;
}

// ---------------------------------------------------------------------------
// Kernel 4: ctx = attn @ V, bf16 result in concat layout [B,S,DIM].
// attn stream (256MB) is the dominant HBM traffic: prefetch next k-tile.
// ---------------------------------------------------------------------------
__global__ __launch_bounds__(256) void ctx_kernel(
    const float* __restrict__ attn, const unsigned short* __restrict__ Vh,
    unsigned short* __restrict__ ctx)
{
    __shared__ alignas(16) unsigned short lA[128 * 64];  // attn tile [q][k] bf16
    __shared__ alignas(16) unsigned short lB[64 * 64];   // V^T tile  [d][k] bf16

    const int bh = blockIdx.z;
    const int q0 = blockIdx.x * 128;
    const int tid = threadIdx.x, lane = tid & 31, wave = tid >> 5;

    const float* A = attn + (size_t)bh * SEQ * SEQ;
    const unsigned short* V = Vh + (size_t)bh * SEQ * DEPTH;

    v8f acc[4] = {};

    for (int k0 = 0; k0 < SEQ; k0 += 64) {
        // attn 128x64 f32 -> bf16 (packed pair stores)
        #pragma unroll
        for (int it = 0; it < 8; ++it) {
            int idx = (it * 256 + tid) * 4;
            int r = idx >> 6, c = idx & 63;
            const float* src = &A[(size_t)(q0 + r) * SEQ + k0 + c];
            if (k0 + 64 < SEQ)
                __builtin_prefetch(src + 64, 0, 1);   // next k-tile -> global_prefetch_b8
            float4 f = *(const float4*)src;
            uint2 pk;
            pk.x = f32x2_to_bf16x2(f.x, f.y);
            pk.y = f32x2_to_bf16x2(f.z, f.w);
            *(uint2*)&lA[r * 64 + c] = pk;
        }
        // V 64(k)x64(d) bf16 -> transpose into lB[d][k]
        #pragma unroll
        for (int it = 0; it < 2; ++it) {
            int idx = (it * 256 + tid) * 8;
            int kk = idx >> 6, dd = idx & 63;
            uint4 raw = *(const uint4*)&V[(size_t)(k0 + kk) * DEPTH + dd];
            const unsigned short* s = (const unsigned short*)&raw;
            #pragma unroll
            for (int j = 0; j < 8; ++j) lB[(dd + j) * 64 + kk] = s[j];
        }
        __syncthreads();

        const int arow = wave * 16 + (lane & 15);
        #pragma unroll
        for (int ks = 0; ks < 64; ks += 32) {
            v16bf a = load_a_frag(lA, 64, arow, ks, lane);
            #pragma unroll
            for (int nt = 0; nt < 4; ++nt) {
                v16bf b = load_b_frag(lB, 64, nt * 16 + (lane & 15), ks, lane);
                acc[nt] = wmma_bf16(a, b, acc[nt]);
            }
        }
        __syncthreads();
    }

    const int bb = bh >> 3, h = bh & 7;
    const int half = lane >> 4;
    #pragma unroll
    for (int nt = 0; nt < 4; ++nt) {
        int d = nt * 16 + (lane & 15);
        #pragma unroll
        for (int i = 0; i < 8; ++i) {
            int ss = q0 + wave * 16 + half * 8 + i;
            ctx[((size_t)bb * SEQ + ss) * DIMSZ + h * DEPTH + d] =
                f32_to_bf16_bits(acc[nt][i]);
        }
    }
}

// ---------------------------------------------------------------------------
// Kernel 5: out = ctx @ Wo + bo, f32 into d_out.
// ---------------------------------------------------------------------------
__global__ __launch_bounds__(256) void out_gemm_kernel(
    const unsigned short* __restrict__ ctx, const float* __restrict__ Wo,
    const float* __restrict__ bo, float* __restrict__ out)
{
    __shared__ alignas(16) unsigned short lA[128 * 64];
    __shared__ alignas(16) unsigned short lB[64 * 64];

    const int m0 = blockIdx.y * 128;
    const int n0 = blockIdx.x * 64;
    const int tid = threadIdx.x, lane = tid & 31, wave = tid >> 5;

    v8f acc[4] = {};

    for (int k0 = 0; k0 < DIMSZ; k0 += 64) {
        // A: ctx already bf16, straight 16B copies
        #pragma unroll
        for (int it = 0; it < 4; ++it) {
            int idx = it * 256 + tid;        // uint4 index; 8 uint4 per 64-short row
            int r = idx >> 3, c8 = idx & 7;
            ((uint4*)lA)[idx] =
                *(const uint4*)&ctx[(size_t)(m0 + r) * DIMSZ + k0 + c8 * 8];
        }
        // B: Wo[k][n] -> lB[n][k]
        #pragma unroll
        for (int it = 0; it < 4; ++it) {
            int idx = (it * 256 + tid) * 4;
            int kk = idx >> 6, nn = idx & 63;
            float4 f = *(const float4*)&Wo[(size_t)(k0 + kk) * DIMSZ + n0 + nn];
            lB[(nn + 0) * 64 + kk] = f32_to_bf16_bits(f.x);
            lB[(nn + 1) * 64 + kk] = f32_to_bf16_bits(f.y);
            lB[(nn + 2) * 64 + kk] = f32_to_bf16_bits(f.z);
            lB[(nn + 3) * 64 + kk] = f32_to_bf16_bits(f.w);
        }
        __syncthreads();

        const int arow = wave * 16 + (lane & 15);
        #pragma unroll
        for (int ks = 0; ks < 64; ks += 32) {
            v16bf a = load_a_frag(lA, 64, arow, ks, lane);
            #pragma unroll
            for (int nt = 0; nt < 4; ++nt) {
                v16bf b = load_b_frag(lB, 64, nt * 16 + (lane & 15), ks, lane);
                acc[nt] = wmma_bf16(a, b, acc[nt]);
            }
        }
        __syncthreads();
    }

    const int half = lane >> 4;
    #pragma unroll
    for (int nt = 0; nt < 4; ++nt) {
        int n = n0 + nt * 16 + (lane & 15);
        float bval = bo[n];
        #pragma unroll
        for (int i = 0; i < 8; ++i) {
            int m = m0 + wave * 16 + half * 8 + i;
            out[(size_t)m * DIMSZ + n] = acc[nt][i] + bval;
        }
    }
}

// ---------------------------------------------------------------------------
extern "C" void kernel_launch(void* const* d_in, const int* in_sizes, int n_in,
                              void* d_out, int out_size, void* d_ws, size_t ws_size,
                              hipStream_t stream) {
    const float* X    = (const float*)d_in[0];
    const float* mask = (const float*)d_in[1];
    const float* Wq   = (const float*)d_in[2];
    const float* bq   = (const float*)d_in[3];
    const float* Wk   = (const float*)d_in[4];
    const float* bk   = (const float*)d_in[5];
    const float* Wv   = (const float*)d_in[6];
    const float* bv   = (const float*)d_in[7];
    const float* Wo   = (const float*)d_in[8];
    const float* bo   = (const float*)d_in[9];

    float* out  = (float*)d_out;                         // [B,S,DIM]
    float* attn = out + (size_t)BATCH * SEQ * DIMSZ;     // [B,H,S,S]

    const size_t headElems = (size_t)BATCH * HEADS * SEQ * DEPTH;  // 2,097,152
    unsigned short* Qh  = (unsigned short*)d_ws;
    unsigned short* Kh  = Qh + headElems;
    unsigned short* Vh  = Kh + headElems;
    unsigned short* ctx = Vh + headElems;                // [B,S,DIM] bf16

    // 1) QKV projections (M=4096, N=512, K=512, x3 weights)
    qkv_gemm_kernel<<<dim3(DIMSZ / 64, (BATCH * SEQ) / 128, 3), 256, 0, stream>>>(
        X, Wq, Wk, Wv, bq, bk, bv, Qh, Kh, Vh);

    // 2) logits (per b,h: M=2048, N=2048, K=64)
    logits_kernel<<<dim3(SEQ / 64, SEQ / 128, BATCH * HEADS), 256, 0, stream>>>(
        Qh, Kh, mask, attn);

    // 3) softmax rows
    softmax_kernel<<<dim3(BATCH * HEADS * SEQ), 256, 0, stream>>>(attn);

    // 4) ctx = attn @ V (per b,h: M=2048, N=64, K=2048)
    ctx_kernel<<<dim3(SEQ / 128, 1, BATCH * HEADS), 256, 0, stream>>>(attn, Vh, ctx);

    // 5) out = ctx @ Wo + bo (M=4096, N=512, K=512)
    out_gemm_kernel<<<dim3(DIMSZ / 64, (BATCH * SEQ) / 128), 256, 0, stream>>>(
        ctx, Wo, bo, out);
}
